// NKIMoEWrapper_88983132438836
// MI455X (gfx1250) — matrix operand order repaired and checked by reading, not compiled, not run on hardware
//
#include <hip/hip_runtime.h>
#include <hip/hip_bf16.h>

typedef __attribute__((ext_vector_type(16))) __bf16 bf16x16;
typedef __attribute__((ext_vector_type(8)))  __bf16 bf16x8;
typedef __attribute__((ext_vector_type(4)))  __bf16 bf16x4;
typedef __attribute__((ext_vector_type(2)))  __bf16 bf16x2;
typedef __attribute__((ext_vector_type(8)))  float  f32x8;
typedef __attribute__((ext_vector_type(4)))  float  f32x4;

#define TOP_K       8
#define NUM_EXPERTS 64
#define HIDDEN      2048
#define INTER       1024
#define NUM_TOKENS  8192
#define BK          32
#define LDP         40   // padded LDS row pitch (elements): 80B, 16B-aligned, conflict-free

// ---------------------------------------------------------------------------
// Router: logits -> softmax -> top-8 -> normalized weights; token 0 -> experts
// ---------------------------------------------------------------------------
__global__ __launch_bounds__(256) void router_topk_kernel(
    const float* __restrict__ hs,   // [T, H]
    const float* __restrict__ rw,   // [H, E]
    float* __restrict__ wts,        // [T, 8]
    int*   __restrict__ rep)        // [8]
{
    __shared__ float red[256];
    __shared__ float pl[NUM_EXPERTS];
    const int t   = blockIdx.x;
    const int tid = threadIdx.x;
    const int e   = tid & 63;
    const int hq  = tid >> 6;            // 0..3, each covers 512 h values
    const float* hrow = hs + (size_t)t * HIDDEN;

    float partial = 0.f;
    const int h0 = hq * (HIDDEN / 4);
    for (int h = 0; h < HIDDEN / 4; ++h)
        partial += hrow[h0 + h] * rw[(size_t)(h0 + h) * NUM_EXPERTS + e];
    red[tid] = partial;
    __syncthreads();
    if (tid < 64)
        pl[e] = red[e] + red[64 + e] + red[128 + e] + red[192 + e];
    __syncthreads();

    if (tid == 0) {
        float mx = pl[0];
        for (int i = 1; i < NUM_EXPERTS; ++i) mx = fmaxf(mx, pl[i]);
        for (int i = 0; i < NUM_EXPERTS; ++i) pl[i] = __expf(pl[i] - mx);
        // softmax denominator cancels after top-k renormalization
        float wv[TOP_K]; int wi[TOP_K]; float s = 0.f;
        for (int slot = 0; slot < TOP_K; ++slot) {
            int bj = 0; float bv = pl[0];
            for (int i = 1; i < NUM_EXPERTS; ++i)
                if (pl[i] > bv) { bv = pl[i]; bj = i; }   // strict >: first index wins ties
            wv[slot] = bv; wi[slot] = bj; s += bv; pl[bj] = -1.f;
        }
        const float inv = 1.f / s;
        for (int slot = 0; slot < TOP_K; ++slot)
            wts[(size_t)t * TOP_K + slot] = wv[slot] * inv;
        if (t == 0)
            for (int slot = 0; slot < TOP_K; ++slot) rep[slot] = wi[slot];
    }
}

// ---------------------------------------------------------------------------
// fp32 -> bf16 bulk convert (hidden_states -> A)
// ---------------------------------------------------------------------------
__global__ __launch_bounds__(256) void cvt_f32_bf16_kernel(
    const float* __restrict__ in, __bf16* __restrict__ out, int n4)
{
    int i = blockIdx.x * blockDim.x + threadIdx.x;
    const int stride = gridDim.x * blockDim.x;
    for (; i < n4; i += stride) {
        f32x4 v = *(const f32x4*)(in + (size_t)i * 4);
        bf16x4 o;
        o[0] = (__bf16)v[0]; o[1] = (__bf16)v[1];
        o[2] = (__bf16)v[2]; o[3] = (__bf16)v[3];
        *(bf16x4*)(out + (size_t)i * 4) = o;
    }
}

// ---------------------------------------------------------------------------
// A-fragment: 16-bit A 16x32 layout (lane = row, two 16B K-runs)
// ---------------------------------------------------------------------------
__device__ __forceinline__ bf16x16 load_a_frag(const __bf16* rowbase, int half)
{
    bf16x8 lo = *(const bf16x8*)(rowbase + half * 8);
    bf16x8 hi = *(const bf16x8*)(rowbase + 16 + half * 8);
    bf16x16 a;
#pragma unroll
    for (int j = 0; j < 8; ++j) { a[j] = lo[j]; a[8 + j] = hi[j]; }
    return a;
}

// B-fragment: 16-bit B 32x16 layout (lane = col, two 16B K-runs)
__device__ __forceinline__ bf16x16 load_b_frag(const __bf16* colbase, int half)
{
    bf16x8 lo = *(const bf16x8*)(colbase + half * 8);
    bf16x8 hi = *(const bf16x8*)(colbase + 16 + half * 8);
    bf16x16 b;
#pragma unroll
    for (int j = 0; j < 8; ++j) { b[j] = lo[j]; b[8 + j] = hi[j]; }
    return b;
}

// pack two fp32 (consecutive k) into one bf16x2 b32 LDS store (v_cvt_pk_bf16_f32)
__device__ __forceinline__ void store_pair(__bf16* dst, float a, float b)
{
    bf16x2 p; p[0] = (__bf16)a; p[1] = (__bf16)b;
    *(bf16x2*)dst = p;
}

// ---------------------------------------------------------------------------
// GEMM1 + SwiGLU: act[T, I] = silu(A@Wg) * (A@Wu), W from gate_up_proj[e]
// block tile: 128 rows x (64 gate cols + matching 64 up cols), 8 waves (4Mx2N)
// LDS ping-pong double buffering: one barrier per K-step; global->reg prefetch
// two tiles ahead overlaps a full iteration of WMMAs.
// ---------------------------------------------------------------------------
__global__ __launch_bounds__(256) void gemm1_swiglu_kernel(
    const __bf16* __restrict__ A,     // [T, H] bf16
    const float*  __restrict__ gup,   // [E, H, 2I] fp32
    const int*    __restrict__ rep,
    int slot,
    __bf16* __restrict__ act)         // [T, I] bf16
{
    __shared__ __bf16 As[2][128 * LDP];
    __shared__ __bf16 Bg[2][64 * LDP];   // transposed [n][k]
    __shared__ __bf16 Bu[2][64 * LDP];

    const int tid  = threadIdx.x;
    const int lane = tid & 31;
    const int wave = tid >> 5;
    const int wm   = wave & 3;        // M subtile: rows wm*32
    const int wn   = wave >> 2;       // N subtile: cols wn*32
    const int half = lane >> 4;
    const int l15  = lane & 15;

    const int m0 = blockIdx.x * 128;
    const int i0 = blockIdx.y * 64;

    // staging thread mapping
    const int c8 = (tid & 3) * 8;     // A: 8-elem chunk in k
    const int rA = tid >> 2;          // A: row 0..63 (and +64)
    const int ci = tid & 15;          // B: float4 chunk in n (16 chunks of 64 cols)
    const int r2 = (tid >> 4) * 2;    // B: even k row 0..30

    const int e = rep[slot];
    const float* B = gup + (size_t)e * HIDDEN * (2 * INTER);

    bf16x8 pa0, pa1;
    f32x4 pg0, pg1, pu0, pu1;

    auto fetch = [&](int kn) {
        pa0 = *(const bf16x8*)(A + (size_t)(m0 + rA) * HIDDEN + kn + c8);
        pa1 = *(const bf16x8*)(A + (size_t)(m0 + rA + 64) * HIDDEN + kn + c8);
        const float* rg = B + (size_t)(kn + r2) * (2 * INTER) + i0 + ci * 4;
        pg0 = *(const f32x4*)(rg);
        pg1 = *(const f32x4*)(rg + 2 * INTER);
        pu0 = *(const f32x4*)(rg + INTER);
        pu1 = *(const f32x4*)(rg + INTER + 2 * INTER);
    };
    auto stage = [&](int b) {
        *(bf16x8*)(&As[b][rA * LDP + c8])        = pa0;
        *(bf16x8*)(&As[b][(rA + 64) * LDP + c8]) = pa1;
#pragma unroll
        for (int j = 0; j < 4; ++j) {
            store_pair(&Bg[b][(ci * 4 + j) * LDP + r2], pg0[j], pg1[j]);
            store_pair(&Bu[b][(ci * 4 + j) * LDP + r2], pu0[j], pu1[j]);
        }
    };

    f32x8 accg[2][2] = {};
    f32x8 accu[2][2] = {};

    const int NT = HIDDEN / BK;   // 64 K-tiles
    fetch(0);
    stage(0);
    fetch(BK);                    // tile 1 in flight across the first compute
    __syncthreads();

    for (int it = 0; it < NT; ++it) {
        const int cur = it & 1;

        bf16x16 af[2], bgf[2], buf2[2];
#pragma unroll
        for (int mi = 0; mi < 2; ++mi)
            af[mi] = load_a_frag(&As[cur][(wm * 32 + mi * 16 + l15) * LDP], half);
#pragma unroll
        for (int ni = 0; ni < 2; ++ni) {
            const int n = wn * 32 + ni * 16 + l15;
            bgf[ni]  = load_b_frag(&Bg[cur][n * LDP], half);
            buf2[ni] = load_b_frag(&Bu[cur][n * LDP], half);
        }
#pragma unroll
        for (int mi = 0; mi < 2; ++mi)
#pragma unroll
            for (int ni = 0; ni < 2; ++ni) {
                accg[mi][ni] = __builtin_amdgcn_wmma_f32_16x16x32_bf16(
                    false, af[mi], false, bgf[ni], (short)0, accg[mi][ni], false, false);
                accu[mi][ni] = __builtin_amdgcn_wmma_f32_16x16x32_bf16(
                    false, af[mi], false, buf2[ni], (short)0, accu[mi][ni], false, false);
            }

        if (it + 1 < NT) {
            stage(cur ^ 1);                         // publish next tile
            if (it + 2 < NT) fetch((it + 2) * BK);  // keep 2 tiles in flight
            __syncthreads();                        // single barrier per K-step
        }
    }

    // fused SwiGLU epilogue -> bf16 act  (fast rcp: single v_rcp_f32)
#pragma unroll
    for (int mi = 0; mi < 2; ++mi)
#pragma unroll
        for (int ni = 0; ni < 2; ++ni) {
            const int col = i0 + wn * 32 + ni * 16 + l15;
#pragma unroll
            for (int v = 0; v < 8; ++v) {
                const int row = m0 + wm * 32 + mi * 16 + v + 8 * half;
                const float g = accg[mi][ni][v];
                const float u = accu[mi][ni][v];
                const float s = g * __builtin_amdgcn_rcpf(1.f + __expf(-g)) * u;
                act[(size_t)row * INTER + col] = (__bf16)s;
            }
        }
}

// ---------------------------------------------------------------------------
// GEMM2: out[T, H] (+)= w[t,slot] * (act @ down_proj[e])
// block tile 128x128, 8 waves (4Mx2N), ping-pong LDS, 2x4 WMMA accumulators
// ---------------------------------------------------------------------------
__global__ __launch_bounds__(256) void gemm2_down_kernel(
    const __bf16* __restrict__ act,   // [T, I] bf16
    const float*  __restrict__ dwn,   // [E, I, H] fp32
    const int*    __restrict__ rep,
    const float*  __restrict__ wts,   // [T, 8]
    int slot,
    float* __restrict__ out,          // [T, H]
    int is_first)
{
    __shared__ __bf16 As[2][128 * LDP];
    __shared__ __bf16 Bs[2][128 * LDP];  // transposed [n][k]
    __shared__ float  wL[128];

    const int tid  = threadIdx.x;
    const int lane = tid & 31;
    const int wave = tid >> 5;
    const int wm   = wave & 3;
    const int wn   = wave >> 2;
    const int half = lane >> 4;
    const int l15  = lane & 15;

    const int m0 = blockIdx.x * 128;
    const int n0 = blockIdx.y * 128;

    const int c8 = (tid & 3) * 8;
    const int rA = tid >> 2;
    const int ci = tid & 31;          // 32 float4 chunks per 128-col row
    const int r2 = (tid >> 5) * 2;    // even k row 0..14 (+16 for second pass)

    const int e = rep[slot];
    const float* B = dwn + (size_t)e * INTER * HIDDEN;

    if (tid < 128) wL[tid] = wts[(size_t)(m0 + tid) * TOP_K + slot];

    bf16x8 pa0, pa1;
    f32x4 pb[4];

    auto fetch = [&](int kn) {
        pa0 = *(const bf16x8*)(act + (size_t)(m0 + rA) * INTER + kn + c8);
        pa1 = *(const bf16x8*)(act + (size_t)(m0 + rA + 64) * INTER + kn + c8);
        const float* rg = B + (size_t)(kn + r2) * HIDDEN + n0 + ci * 4;
        pb[0] = *(const f32x4*)(rg);
        pb[1] = *(const f32x4*)(rg + HIDDEN);
        pb[2] = *(const f32x4*)(rg + 16 * HIDDEN);
        pb[3] = *(const f32x4*)(rg + 17 * HIDDEN);
    };
    auto stage = [&](int b) {
        *(bf16x8*)(&As[b][rA * LDP + c8])        = pa0;
        *(bf16x8*)(&As[b][(rA + 64) * LDP + c8]) = pa1;
#pragma unroll
        for (int p = 0; p < 2; ++p)
#pragma unroll
            for (int j = 0; j < 4; ++j)
                store_pair(&Bs[b][(ci * 4 + j) * LDP + r2 + p * 16],
                           pb[2 * p][j], pb[2 * p + 1][j]);
    };

    f32x8 acc[2][4] = {};

    const int NT = INTER / BK;    // 32 K-tiles
    fetch(0);
    stage(0);
    fetch(BK);
    __syncthreads();

    for (int it = 0; it < NT; ++it) {
        const int cur = it & 1;

        bf16x16 af[2], bfr[4];
#pragma unroll
        for (int mi = 0; mi < 2; ++mi)
            af[mi] = load_a_frag(&As[cur][(wm * 32 + mi * 16 + l15) * LDP], half);
#pragma unroll
        for (int ni = 0; ni < 4; ++ni)
            bfr[ni] = load_b_frag(&Bs[cur][(wn * 64 + ni * 16 + l15) * LDP], half);
#pragma unroll
        for (int mi = 0; mi < 2; ++mi)
#pragma unroll
            for (int ni = 0; ni < 4; ++ni)
                acc[mi][ni] = __builtin_amdgcn_wmma_f32_16x16x32_bf16(
                    false, af[mi], false, bfr[ni], (short)0, acc[mi][ni], false, false);

        if (it + 1 < NT) {
            stage(cur ^ 1);
            if (it + 2 < NT) fetch((it + 2) * BK);
            __syncthreads();
        }
    }

    // weighted accumulate into out (kernels serialized per slot -> deterministic)
#pragma unroll
    for (int mi = 0; mi < 2; ++mi)
#pragma unroll
        for (int ni = 0; ni < 4; ++ni) {
            const int col = n0 + wn * 64 + ni * 16 + l15;
#pragma unroll
            for (int v = 0; v < 8; ++v) {
                const int rloc = wm * 32 + mi * 16 + v + 8 * half;
                const size_t idx = (size_t)(m0 + rloc) * HIDDEN + col;
                const float w = wL[rloc];
                const float prev = is_first ? 0.f : out[idx];
                out[idx] = prev + w * acc[mi][ni][v];
            }
        }
}

// ---------------------------------------------------------------------------
extern "C" void kernel_launch(void* const* d_in, const int* in_sizes, int n_in,
                              void* d_out, int out_size, void* d_ws, size_t ws_size,
                              hipStream_t stream)
{
    const float* hs  = (const float*)d_in[0];   // [T, H]
    const float* rw  = (const float*)d_in[1];   // [H, E]
    const float* gup = (const float*)d_in[2];   // [E, H, 2I]
    const float* dwn = (const float*)d_in[3];   // [E, I, H]
    float* out = (float*)d_out;

    char* ws = (char*)d_ws;
    float*  wts = (float*)(ws);                                     // 256 KB
    int*    rep = (int*)(ws + 256 * 1024);                          // 32 B
    __bf16* A   = (__bf16*)(ws + 512 * 1024);                       // 32 MB
    __bf16* act = (__bf16*)(ws + 512 * 1024 + 32ull * 1024 * 1024); // 16 MB

    router_topk_kernel<<<NUM_TOKENS, 256, 0, stream>>>(hs, rw, wts, rep);
    cvt_f32_bf16_kernel<<<4096, 256, 0, stream>>>(hs, A, NUM_TOKENS * HIDDEN / 4);

    for (int k = 0; k < TOP_K; ++k) {
        gemm1_swiglu_kernel<<<dim3(NUM_TOKENS / 128, INTER / 64), 256, 0, stream>>>(
            A, gup, rep, k, act);
        gemm2_down_kernel<<<dim3(NUM_TOKENS / 128, HIDDEN / 128), 256, 0, stream>>>(
            act, dwn, rep, wts, k, out, k == 0);
    }
}
